// PhiAttention_22643067585102
// MI455X (gfx1250) — matrix operand order repaired and checked by reading, not compiled, not run on hardware
//
#include <hip/hip_runtime.h>
#include <hip/hip_bf16.h>

// ---------------------------------------------------------------------------
// PhiAttention for gfx1250 (MI455X, wave32, WMMA).
// Pipeline: pack f32->f16 (x3) -> qkv GEMM -> RoPE -> flash attention -> out GEMM.
// Matmuls: v_wmma_f32_16x16x32_f16 (f16 in, f32 accum), LDS double-buffered
// with GLOBAL_LOAD_ASYNC_TO_LDS_B128 staging (ASYNCcnt-tracked).
// ---------------------------------------------------------------------------

#define HIDDEN    2560
#define NUM_HEADS 32
#define HEAD      80
#define ROT       32
#define BATCH     2
#define SEQ       2048
#define QKV_DIM   (3 * HIDDEN)   // 7680
#define BS        (BATCH * SEQ)  // 4096

typedef _Float16     v16h  __attribute__((ext_vector_type(16)));
typedef float        v8f   __attribute__((ext_vector_type(8)));
typedef unsigned int u32x4 __attribute__((ext_vector_type(4)));
typedef unsigned int u32x2 __attribute__((ext_vector_type(2)));
typedef int          i32x4v __attribute__((vector_size(16)));  // builtin's pointee type

union Frag {            // one WMMA A/B operand: 16 halves = 8 VGPRs
    v16h  h;
    u32x4 q[2];
};
union H4 {              // 4 packed halves for 8-byte stores
    _Float16 h[4];
    u32x2    u;
};

#define WMMA_F16(A, B, C) \
    __builtin_amdgcn_wmma_f32_16x16x32_f16(false, (A), false, (B), (short)0, (C), false, false)

// ---- async global->LDS staging (CDNA5), with synchronous fallback ---------
#if defined(__has_builtin)
#if __has_builtin(__builtin_amdgcn_global_load_async_to_lds_b128) && \
    __has_builtin(__builtin_amdgcn_s_wait_asynccnt)
#define HAVE_ASYNC_LDS 1
#endif
#endif

#define AS1 __attribute__((address_space(1)))
#define AS3 __attribute__((address_space(3)))

__device__ __forceinline__ void copy16_to_lds(_Float16* ldsDst, const _Float16* gSrc)
{
#ifdef HAVE_ASYNC_LDS
    __builtin_amdgcn_global_load_async_to_lds_b128(
        (AS1 i32x4v*)gSrc, (AS3 i32x4v*)ldsDst, 0, 0);
#else
    *reinterpret_cast<u32x4*>(ldsDst) = *reinterpret_cast<const u32x4*>(gSrc);
#endif
}
__device__ __forceinline__ void async_wait_all()
{
#ifdef HAVE_ASYNC_LDS
    __builtin_amdgcn_s_wait_asynccnt(0);
#endif
}
__device__ __forceinline__ void async_wait_group4()
{
#ifdef HAVE_ASYNC_LDS
    __builtin_amdgcn_s_wait_asynccnt(4);   // allow newest group (4 ops) in flight
#endif
}

// ===========================================================================
// Kernel 0: pack f32 -> f16 (n multiple of 4), float4 -> 8B stores.
// ===========================================================================
__global__ __launch_bounds__(256)
void pack_f16_kernel(const float* __restrict__ src, _Float16* __restrict__ dst, int n4)
{
    const int idx = blockIdx.x * blockDim.x + threadIdx.x;
    if (idx >= n4) return;
    const float4 v = *reinterpret_cast<const float4*>(&src[(size_t)idx * 4]);
    H4 h; h.h[0] = (_Float16)v.x; h.h[1] = (_Float16)v.y;
          h.h[2] = (_Float16)v.z; h.h[3] = (_Float16)v.w;
    *reinterpret_cast<u32x2*>(&dst[(size_t)idx * 4]) = h.u;
}

// ===========================================================================
// Kernel 1/2: generic NT GEMM  C[M,N] = A[M,K] * W[N,K]^T + bias[N]
//   A, W row-major f16.  Block: 256 threads (8 waves), tile 128x128, K-step 32.
//   LDS double-buffered; staging is a pure async 16B copy.
//   Waves 4(M) x 2(N); each wave: 2 A-frags x 4 B-frags -> 8 accums.
// ===========================================================================
#define GT_M 128
#define GT_N 128
#define GT_K 32
#define LDA  40   // LDS row stride in halves (80B rows, 16B aligned)

template <typename OutT>
__global__ __launch_bounds__(256)
void gemm_nt_wmma_f16(const _Float16* __restrict__ A, const _Float16* __restrict__ W,
                      const float* __restrict__ bias, OutT* __restrict__ C,
                      int M, int N, int K)
{
    __shared__ __align__(16) _Float16 As[2][GT_M * LDA];
    __shared__ __align__(16) _Float16 Bs[2][GT_N * LDA];

    const int tid   = threadIdx.x;
    const int wave  = tid >> 5;
    const int lane  = tid & 31;
    const int lhalf = lane >> 4;
    const int lm    = lane & 15;
    const int waveM = wave >> 1;      // 0..3
    const int waveN = wave & 1;       // 0..1

    const int tileM = blockIdx.y * GT_M;
    const int tileN = blockIdx.x * GT_N;

    v8f acc[2][4];
#pragma unroll
    for (int i = 0; i < 2; ++i)
#pragma unroll
        for (int j = 0; j < 4; ++j)
#pragma unroll
            for (int e = 0; e < 8; ++e) acc[i][j][e] = 0.0f;

    // Stage one 128x32 f16 tile pair: 512 chunks of 16B each per tile,
    // 256 threads x 2 chunks; each thread issues 4 async copies total.
    auto stage = [&](int kt, int bufi) {
#pragma unroll
        for (int i = 0; i < 2; ++i) {
            const int c   = tid + i * 256;
            const int row = c >> 2;
            const int col = (c & 3) * 8;       // halves
            copy16_to_lds(&As[bufi][row * LDA + col],
                          &A[(size_t)(tileM + row) * K + kt * GT_K + col]);
            copy16_to_lds(&Bs[bufi][row * LDA + col],
                          &W[(size_t)(tileN + row) * K + kt * GT_K + col]);
        }
    };

    const int nK = K / GT_K;
    stage(0, 0);
    for (int kt = 0; kt < nK; ++kt) {
        const int cur = kt & 1;
        if (kt + 1 < nK) {
            __syncthreads();              // everyone done reading buf cur^1
            stage(kt + 1, cur ^ 1);
            async_wait_group4();          // buf `cur` (older group) complete
        } else {
            async_wait_all();
        }
        __syncthreads();                  // buf `cur` visible to all waves

        // A-operand: lane row M=lm; slots 0-7 -> K base..base+7,
        // slots 8-15 -> K base+16..+23, base = 8*lhalf.
        Frag af[2];
#pragma unroll
        for (int i = 0; i < 2; ++i) {
            const int off = (waveM * 32 + i * 16 + lm) * LDA + lhalf * 8;
            af[i].q[0] = *reinterpret_cast<const u32x4*>(&As[cur][off]);
            af[i].q[1] = *reinterpret_cast<const u32x4*>(&As[cur][off + 16]);
        }
        // B-operand: lane col N=lm; same K slots as A (8*lhalf, +16).
        Frag bf[4];
#pragma unroll
        for (int j = 0; j < 4; ++j) {
            const int off = (waveN * 64 + j * 16 + lm) * LDA + lhalf * 8;
            bf[j].q[0] = *reinterpret_cast<const u32x4*>(&Bs[cur][off]);
            bf[j].q[1] = *reinterpret_cast<const u32x4*>(&Bs[cur][off + 16]);
        }
#pragma unroll
        for (int i = 0; i < 2; ++i)
#pragma unroll
            for (int j = 0; j < 4; ++j)
                acc[i][j] = WMMA_F16(af[i].h, bf[j].h, acc[i][j]);
    }

    // C/D layout: VGPR g, lanes 0-15 -> (M=g, N=lane); lanes 16-31 -> (M=g+8).
#pragma unroll
    for (int i = 0; i < 2; ++i) {
        const int rBase = tileM + waveM * 32 + i * 16 + lhalf * 8;
#pragma unroll
        for (int j = 0; j < 4; ++j) {
            const int cn = tileN + waveN * 64 + j * 16 + lm;
            const float bv = bias[cn];
#pragma unroll
            for (int g = 0; g < 8; ++g)
                C[(size_t)(rBase + g) * N + cn] = (OutT)(acc[i][j][g] + bv);
        }
    }
}

// ===========================================================================
// Kernel 3: Neox RoPE on first ROT dims of q and k (in-place on f16 qkv).
// ===========================================================================
__global__ __launch_bounds__(256)
void rope_kernel(const int* __restrict__ pos_ids, _Float16* __restrict__ qkv)
{
    const int idx = blockIdx.x * blockDim.x + threadIdx.x;
    if (idx >= BS * NUM_HEADS * 16) return;
    const int i  = idx & 15;          // rotation pair index
    const int h  = (idx >> 4) & 31;
    const int bs = idx >> 9;          // flattened (b, s)

    const float p   = (float)pos_ids[bs];
    const float inv = __powf(10000.0f, -(float)i * (1.0f / 16.0f));
    float s, c;
    __sincosf(p * inv, &s, &c);

    size_t base = (size_t)bs * QKV_DIM + h * HEAD;   // q slice
#pragma unroll
    for (int part = 0; part < 2; ++part) {           // q then k
        const float x1 = (float)qkv[base + i];
        const float x2 = (float)qkv[base + i + 16];
        qkv[base + i]      = (_Float16)(x1 * c - x2 * s);
        qkv[base + i + 16] = (_Float16)(x2 * c + x1 * s);
        base += HIDDEN;                              // -> k slice
    }
}

// ===========================================================================
// Kernel 4: causal flash attention, one block = 64 queries of one (b,h).
//   128 threads (4 waves); wave w owns query rows 16w..16w+15.
//   Scores: Q[16x96] x K^T (D padded 80->96, 3 WMMA K-steps), scale applied
//   post-WMMA so Q/K staging is a pure async copy.
//   PV: P[16x64] x V[64x80] (2 WMMA K-steps, V staged transposed).
//   All fragments are preloaded into register arrays before each WMMA burst
//   so the LDS loads batch into clauses with staggered dscnt waits.
// ===========================================================================
#define QB 64
#define KB 64
#define LQ 104    // Qs/Ks LDS stride in halves (208B, 16B aligned)
#define LV 72     // Vt/Ps LDS stride in halves (144B, 16B aligned)

__global__ __launch_bounds__(128)
void attn_kernel(const _Float16* __restrict__ qkv, _Float16* __restrict__ ctx)
{
    __shared__ __align__(16) _Float16 Qs[QB * LQ];        // 13312 B
    __shared__ __align__(16) _Float16 Ks[KB * LQ];        // 13312 B
    __shared__ __align__(16) _Float16 Vt[HEAD * LV];      // 11520 B (transposed)
    __shared__ __align__(16) _Float16 Ps[4 * 16 * LV];    //  9216 B (per wave)

    const int tid   = threadIdx.x;
    const int wave  = tid >> 5;
    const int lane  = tid & 31;
    const int lhalf = lane >> 4;
    const int lm    = lane & 15;

    const int qBase = blockIdx.x * QB;
    const int b     = blockIdx.y >> 5;
    const int h     = blockIdx.y & 31;

    const float scale = 0.11180339887498948f;  // 80^-0.5

    const _Float16* qPtr = qkv + (size_t)(b * SEQ) * QKV_DIM + h * HEAD;
    const _Float16* kPtr = qPtr + HIDDEN;
    const _Float16* vPtr = qPtr + 2 * HIDDEN;

    // Zero the pad columns (d = 80..95) of Qs/Ks once.
    for (int idx = tid; idx < QB * 16; idx += 128) {
        const int r = idx >> 4, d = HEAD + (idx & 15);
        Qs[r * LQ + d] = (_Float16)0.0f;
        Ks[r * LQ + d] = (_Float16)0.0f;
    }
    // Stage Q tile once: 64 rows x 10 16B-chunks = 640 chunks, 5 per thread.
#pragma unroll
    for (int i = 0; i < 5; ++i) {
        const int c = tid + i * 128;
        const int r = c / 10, x = (c - r * 10) * 8;   // halves
        copy16_to_lds(&Qs[r * LQ + x],
                      &qPtr[(size_t)(qBase + r) * QKV_DIM + x]);
    }

    v8f ctxAcc[5];
#pragma unroll
    for (int t = 0; t < 5; ++t)
#pragma unroll
        for (int e = 0; e < 8; ++e) ctxAcc[t][e] = 0.0f;
    float mrow[8], lrow[8];
#pragma unroll
    for (int g = 0; g < 8; ++g) { mrow[g] = -1e30f; lrow[g] = 0.0f; }

    const int nKB = qBase / KB + 1;   // causal: key blocks 0..diag inclusive
    for (int kb = 0; kb < nKB; ++kb) {
        const int kBase = kb * KB;
        __syncthreads();              // previous block's reads of Ks/Vt done
        // Stage K tile (async copy) and V transposed (scalar f16).
#pragma unroll
        for (int i = 0; i < 5; ++i) {
            const int c = tid + i * 128;
            const int r = c / 10, x = (c - r * 10) * 8;
            copy16_to_lds(&Ks[r * LQ + x],
                          &kPtr[(size_t)(kBase + r) * QKV_DIM + x]);
        }
        for (int idx = tid; idx < KB * HEAD; idx += 128) {
            const int r = idx / HEAD, d = idx - r * HEAD;
            Vt[d * LV + r] = vPtr[(size_t)(kBase + r) * QKV_DIM + d];
        }
        async_wait_all();
        __syncthreads();

        // ---- scores: 16 queries x 64 keys per wave ----
        v8f sc[4];
#pragma unroll
        for (int j = 0; j < 4; ++j)
#pragma unroll
            for (int e = 0; e < 8; ++e) sc[j][e] = 0.0f;
#pragma unroll
        for (int kk = 0; kk < 3; ++kk) {
            // Preload 1 Q-frag + 4 K-frags, then issue 4 back-to-back WMMAs.
            Frag aq;
            {
                const int off = (wave * 16 + lm) * LQ + kk * 32 + lhalf * 8;
                aq.q[0] = *reinterpret_cast<const u32x4*>(&Qs[off]);
                aq.q[1] = *reinterpret_cast<const u32x4*>(&Qs[off + 16]);
            }
            Frag bk[4];
#pragma unroll
            for (int j = 0; j < 4; ++j) {
                const int off = (j * 16 + lm) * LQ + kk * 32 + lhalf * 8;
                bk[j].q[0] = *reinterpret_cast<const u32x4*>(&Ks[off]);
                bk[j].q[1] = *reinterpret_cast<const u32x4*>(&Ks[off + 16]);
            }
#pragma unroll
            for (int j = 0; j < 4; ++j)
                sc[j] = WMMA_F16(aq.h, bk[j].h, sc[j]);
        }

        // ---- scale + causal mask + online softmax (rows = g + 8*lhalf) ----
        float corr[8];
#pragma unroll
        for (int g = 0; g < 8; ++g) {
            const int q = qBase + wave * 16 + g + 8 * lhalf;
            float rmax = -1e30f;
#pragma unroll
            for (int j = 0; j < 4; ++j) {
                const int key = kBase + j * 16 + lm;
                const float sv = (key <= q) ? sc[j][g] * scale : -1e30f;
                sc[j][g] = sv;
                rmax = fmaxf(rmax, sv);
            }
#pragma unroll
            for (int m = 1; m < 16; m <<= 1)
                rmax = fmaxf(rmax, __shfl_xor(rmax, m, 32));
            const float mnew = fmaxf(mrow[g], rmax);
            const float cr   = __expf(mrow[g] - mnew);
            mrow[g] = mnew;
            float rsum = 0.0f;
#pragma unroll
            for (int j = 0; j < 4; ++j) {
                const float pv = __expf(sc[j][g] - mnew);
                sc[j][g] = pv;
                rsum += pv;
            }
#pragma unroll
            for (int m = 1; m < 16; m <<= 1)
                rsum += __shfl_xor(rsum, m, 32);
            lrow[g] = lrow[g] * cr + rsum;
            corr[g] = cr;
        }
#pragma unroll
        for (int t = 0; t < 5; ++t)
#pragma unroll
            for (int g = 0; g < 8; ++g) ctxAcc[t][g] *= corr[g];

        // ---- P: C-layout regs -> f16 A-layout via per-wave LDS ----
        _Float16* Pw = &Ps[wave * 16 * LV];
#pragma unroll
        for (int j = 0; j < 4; ++j)
#pragma unroll
            for (int g = 0; g < 8; ++g)
                Pw[(g + 8 * lhalf) * LV + j * 16 + lm] = (_Float16)sc[j][g];
        __syncthreads();

        // ---- PV: ctx[16x80] += P[16x64] * V[64x80] ----
#pragma unroll
        for (int kk = 0; kk < 2; ++kk) {
            // Preload 1 P-frag + 5 V-frags, then issue 5 back-to-back WMMAs.
            Frag ap;
            {
                const int off = lm * LV + kk * 32 + lhalf * 8;
                ap.q[0] = *reinterpret_cast<const u32x4*>(&Pw[off]);
                ap.q[1] = *reinterpret_cast<const u32x4*>(&Pw[off + 16]);
            }
            Frag bv[5];
#pragma unroll
            for (int t = 0; t < 5; ++t) {
                const int off = (t * 16 + lm) * LV + kk * 32 + lhalf * 8;
                bv[t].q[0] = *reinterpret_cast<const u32x4*>(&Vt[off]);
                bv[t].q[1] = *reinterpret_cast<const u32x4*>(&Vt[off + 16]);
            }
#pragma unroll
            for (int t = 0; t < 5; ++t)
                ctxAcc[t] = WMMA_F16(ap.h, bv[t].h, ctxAcc[t]);
        }
    }

    // ---- epilogue: (ctx / l) -> f16 [B,S,HIDDEN] ----
    _Float16* outBase = ctx + (size_t)(b * SEQ) * HIDDEN + h * HEAD;
#pragma unroll
    for (int t = 0; t < 5; ++t) {
        const int d = t * 16 + lm;
#pragma unroll
        for (int g = 0; g < 8; ++g) {
            const int q = qBase + wave * 16 + g + 8 * lhalf;
            outBase[(size_t)q * HIDDEN + d] = (_Float16)(ctxAcc[t][g] / lrow[g]);
        }
    }
}

// ===========================================================================
// Launch. Workspace layout (f16 halves):
//   qkv16   [4096,7680]  @ 0
//   ctx16   [4096,2560]
//   hid16   [4096,2560]
//   wqkv16  [7680,2560]
//   outw16  [2560,2560]      total ~150 MiB
// ===========================================================================
extern "C" void kernel_launch(void* const* d_in, const int* in_sizes, int n_in,
                              void* d_out, int out_size, void* d_ws, size_t ws_size,
                              hipStream_t stream)
{
    (void)in_sizes; (void)n_in; (void)out_size; (void)ws_size;
    const int*   pos    = (const int*)d_in[0];
    const float* hidden = (const float*)d_in[1];
    const float* wqkv   = (const float*)d_in[2];
    const float* bqkv   = (const float*)d_in[3];
    const float* wout   = (const float*)d_in[4];
    const float* bout   = (const float*)d_in[5];
    float*       out    = (float*)d_out;

    _Float16* qkv16  = (_Float16*)d_ws;
    _Float16* ctx16  = qkv16  + (size_t)BS * QKV_DIM;
    _Float16* hid16  = ctx16  + (size_t)BS * HIDDEN;
    _Float16* wqkv16 = hid16  + (size_t)BS * HIDDEN;
    _Float16* outw16 = wqkv16 + (size_t)QKV_DIM * HIDDEN;

    // Pack the three f32 operands to f16 (bandwidth-bound, one pass each).
    {
        const int n4h = BS * HIDDEN / 4;
        pack_f16_kernel<<<(n4h + 255) / 256, 256, 0, stream>>>(hidden, hid16, n4h);
        const int n4w = QKV_DIM * HIDDEN / 4;
        pack_f16_kernel<<<(n4w + 255) / 256, 256, 0, stream>>>(wqkv, wqkv16, n4w);
        const int n4o = HIDDEN * HIDDEN / 4;
        pack_f16_kernel<<<(n4o + 255) / 256, 256, 0, stream>>>(wout, outw16, n4o);
    }

    dim3 g1(QKV_DIM / GT_N, BS / GT_M);
    gemm_nt_wmma_f16<_Float16><<<g1, 256, 0, stream>>>(hid16, wqkv16, bqkv, qkv16,
                                                       BS, QKV_DIM, HIDDEN);

    rope_kernel<<<(BS * NUM_HEADS * 16) / 256, 256, 0, stream>>>(pos, qkv16);

    dim3 g2(SEQ / QB, BATCH * NUM_HEADS);
    attn_kernel<<<g2, 128, 0, stream>>>(qkv16, ctx16);

    dim3 g3(HIDDEN / GT_N, BS / GT_M);
    gemm_nt_wmma_f16<float><<<g3, 256, 0, stream>>>(ctx16, outw16, bout, out,
                                                    BS, HIDDEN, HIDDEN);
}
